// MultiHeadedAttention_23974507446409
// MI455X (gfx1250) — matrix operand order repaired and checked by reading, not compiled, not run on hardware
//
#include <hip/hip_runtime.h>
#include <hip/hip_bf16.h>

typedef __bf16 bf16;
typedef __attribute__((ext_vector_type(16))) __bf16 v16bf;
typedef __attribute__((ext_vector_type(8)))  __bf16 v8bf;
typedef __attribute__((ext_vector_type(4)))  __bf16 v4bf;
typedef __attribute__((ext_vector_type(8)))  float  v8f;
typedef __attribute__((ext_vector_type(4)))  unsigned int v4u;
typedef __attribute__((ext_vector_type(8)))  int v8i;
typedef __attribute__((ext_vector_type(4)))  int v4i;

#define BATCH  4
#define SEQ    2048
#define DMODEL 1024
#define NH     16
#define DK     64

__device__ __forceinline__ bf16 f2bf(float f) {
  unsigned u = __builtin_bit_cast(unsigned, f);
  unsigned r = u + 0x7FFFu + ((u >> 16) & 1u);
  unsigned short hs = (unsigned short)(r >> 16);
  return __builtin_bit_cast(bf16, hs);
}

__device__ __forceinline__ v16bf pack16(v8bf lo, v8bf hi) {
  v16bf x;
#pragma unroll
  for (int j = 0; j < 8; ++j) { x[j] = lo[j]; x[8 + j] = hi[j]; }
  return x;
}

// ---------------------------------------------------------------- TDM 2D tile load
// Builds a Tensor DMA descriptor (D#) per CDNA5 ISA and issues
// TENSOR_LOAD_TO_LDS (6-arg clang-23 form: g0,g1,g2,g3,g4,cpol). 2D tile: tileD1 rows
// x tileD0 elements (bf16, data_size=2B), source row stride strideD0 elements.
// LDS padding: (padAmount+1) DWORDs inserted every 2^(padInterval+1) DWORDs ->
// reproduces the banked LDS row strides used below. Tracked by TENSORcnt; EXEC ignored.
__device__ __forceinline__ void tdm_load_2d(unsigned ldsAddr, const bf16* gptr,
                                            unsigned tileD0, unsigned tileD1,
                                            unsigned tensD0, unsigned tensD1,
                                            unsigned strideD0,
                                            unsigned padInterval, unsigned padAmount) {
  unsigned long long ga = (unsigned long long)(const void*)gptr;
  v4u g0;
  g0[0] = 1u;                                             // count=1, user mode
  g0[1] = ldsAddr;                                        // lds_addr (bytes)
  g0[2] = (unsigned)ga;                                   // global_addr[31:0]
  g0[3] = (unsigned)((ga >> 32) & 0x01FFFFFFu) | (2u << 30);  // global_addr[56:32] | type=2
  v8i g1;
  g1[0] = (int)((1u << 16) | (1u << 20) |                 // data_size=2B | pad_enable
                (padInterval << 22) | (padAmount << 25));
  g1[1] = (int)((tensD0 & 0xFFFFu) << 16);                // tensor_dim0[15:0]
  g1[2] = (int)((tensD0 >> 16) | ((tensD1 & 0xFFFFu) << 16));  // tdim0[31:16] | tdim1[15:0]
  g1[3] = (int)((tensD1 >> 16) | (tileD0 << 16));         // tdim1[31:16] | tile_dim0
  g1[4] = (int)(tileD1 & 0xFFFFu);                        // tile_dim1 (tile_dim2=0)
  g1[5] = (int)strideD0;                                  // tensor_dim0_stride[31:0]
  g1[6] = 0;
  g1[7] = 0;
  v4i z4 = {};
  v8i z8 = {};
  __builtin_amdgcn_tensor_load_to_lds(g0, g1, z4, z4, z8, 0);
}

// ---------------------------------------------------------------- fp32 -> bf16
__global__ void cvt_f32_bf16(const float* __restrict__ src, bf16* __restrict__ dst, int n4) {
  int i = blockIdx.x * blockDim.x + threadIdx.x;
  int stride = gridDim.x * blockDim.x;
  for (; i < n4; i += stride) {
    float4 v = ((const float4*)src)[i];
    v4bf o;
    o[0] = f2bf(v.x); o[1] = f2bf(v.y); o[2] = f2bf(v.z); o[3] = f2bf(v.w);
    ((v4bf*)dst)[i] = o;
  }
}

// ---------------------------------------------------------------- Y = A * W^T
// A: [M,K] bf16 row-major, W: [N,K] bf16 row-major (torch Linear weight layout).
// Block tile 128x128, BK=32, 256 threads (8 waves), wave tile 32x64.
// Double-buffered LDS; tiles DMA'd by the Tensor Data Mover (wave 0 issues),
// so slab k+1 streams in while slab k runs on the matrix cores.
template <bool OUT_F32>
__global__ __launch_bounds__(256)
void gemm_xwT(const bf16* __restrict__ A, const bf16* __restrict__ W,
              float* __restrict__ Of, bf16* __restrict__ Ob,
              int M, int N, int K) {
  // row stride 40 elems (80B): 16B-aligned fragment reads, conflict-skewed.
  // TDM pad: every 16 DWORDs (=32 bf16 row) insert 4 DWORDs -> stride 40. encode (3,3).
  __shared__ alignas(16) bf16 As[2][128][40];
  __shared__ alignas(16) bf16 Bs[2][128][40];

  const int tid   = threadIdx.x;
  const int lane  = tid & 31;
  const int wave  = tid >> 5;
  const int h     = lane >> 4;   // half of wave
  const int r     = lane & 15;
  const int waveM = wave >> 1;   // 0..3
  const int waveN = wave & 1;    // 0..1
  const int row0  = blockIdx.y * 128;
  const int col0  = blockIdx.x * 128;

  v8f acc[2][4] = {};

  const int nk = K / 32;
  if (wave == 0) {
    tdm_load_2d((unsigned)(unsigned long long)(void*)&As[0][0][0],
                A + (size_t)row0 * K, 32, 128, (unsigned)K, (unsigned)(M - row0),
                (unsigned)K, 3, 3);
    tdm_load_2d((unsigned)(unsigned long long)(void*)&Bs[0][0][0],
                W + (size_t)col0 * K, 32, 128, (unsigned)K, (unsigned)(N - col0),
                (unsigned)K, 3, 3);
  }
  __builtin_amdgcn_s_wait_tensorcnt(0);
  __syncthreads();

  for (int k = 0; k < nk; ++k) {
    const int cur = k & 1;
    const int nxt = cur ^ 1;
    if ((k + 1 < nk) && wave == 0) {  // DMA next slab while we compute this one
      tdm_load_2d((unsigned)(unsigned long long)(void*)&As[nxt][0][0],
                  A + (size_t)row0 * K + (k + 1) * 32, 32, 128,
                  (unsigned)K, (unsigned)(M - row0), (unsigned)K, 3, 3);
      tdm_load_2d((unsigned)(unsigned long long)(void*)&Bs[nxt][0][0],
                  W + (size_t)col0 * K + (k + 1) * 32, 32, 128,
                  (unsigned)K, (unsigned)(N - col0), (unsigned)K, 3, 3);
    }

    v16bf afrag[2];
#pragma unroll
    for (int mi = 0; mi < 2; ++mi) {
      int arow = waveM * 32 + mi * 16 + r;
      afrag[mi] = pack16(*(const v8bf*)(&As[cur][arow][8 * h]),
                         *(const v8bf*)(&As[cur][arow][16 + 8 * h]));
    }
#pragma unroll
    for (int ni = 0; ni < 4; ++ni) {
      int bcol = waveN * 64 + ni * 16 + r;
      v16bf bfrag = pack16(*(const v8bf*)(&Bs[cur][bcol][8 * h]),
                           *(const v8bf*)(&Bs[cur][bcol][16 + 8 * h]));
#pragma unroll
      for (int mi = 0; mi < 2; ++mi)
        acc[mi][ni] = __builtin_amdgcn_wmma_f32_16x16x32_bf16(
            false, afrag[mi], false, bfrag, (short)0, acc[mi][ni], false, false);
    }

    __builtin_amdgcn_s_wait_tensorcnt(0);  // next-slab DMA landed (no-op for waves 1..7)
    __syncthreads();
  }

#pragma unroll
  for (int mi = 0; mi < 2; ++mi)
#pragma unroll
    for (int ni = 0; ni < 4; ++ni) {
      int colg = col0 + waveN * 64 + ni * 16 + r;
#pragma unroll
      for (int i = 0; i < 8; ++i) {
        int rowg = row0 + waveM * 32 + mi * 16 + 8 * h + i;
        float v = acc[mi][ni][i];
        if (OUT_F32) Of[(size_t)rowg * N + colg] = v;
        else         Ob[(size_t)rowg * N + colg] = f2bf(v);
      }
    }
}

// ---------------------------------------------------------------- flash attention
// One block = 64 q rows of one (b,h). 4 waves; wave w owns q rows [16w,16w+16).
// Q,K,V,O are [B,S,D] bf16; head slice at column h*DK. Q/K tiles come in via TDM
// (pad encode (4,3): 4 DWORDs every 32 -> row stride 72); V is transposed manually.
__global__ __launch_bounds__(128)
void flash_attn(const bf16* __restrict__ Q, const bf16* __restrict__ K,
                const bf16* __restrict__ V, bf16* __restrict__ O) {
  __shared__ alignas(16) bf16 Qs[64][72];
  __shared__ alignas(16) bf16 Ks[64][72];
  __shared__ alignas(16) bf16 Vt[64][72];  // transposed: Vt[dk][kv]
  __shared__ alignas(16) bf16 Ps[64][72];  // wave-private 16-row strips

  const int tid  = threadIdx.x;
  const int lane = tid & 31;
  const int wave = tid >> 5;  // 0..3
  const int h    = lane >> 4;
  const int r    = lane & 15;

  const int q0 = blockIdx.x * 64;
  const int b  = blockIdx.y / NH;
  const int hh = blockIdx.y % NH;

  const size_t base = (size_t)b * SEQ * DMODEL + (size_t)hh * DK;
  const bf16* Qg = Q + base;
  const bf16* Kg = K + base;
  const bf16* Vg = V + base;
  bf16*       Og = O + base;

  if (wave == 0)  // Q tile (64x64) via Tensor Data Mover
    tdm_load_2d((unsigned)(unsigned long long)(void*)&Qs[0][0],
                Qg + (size_t)q0 * DMODEL, 64, 64, DK, (unsigned)(SEQ - q0),
                DMODEL, 4, 3);

  float rowm[8], rowl[8];
#pragma unroll
  for (int i = 0; i < 8; ++i) { rowm[i] = -1e30f; rowl[i] = 0.f; }
  v8f accO[4] = {};

  for (int kv0 = 0; kv0 < SEQ; kv0 += 64) {
    if (wave == 0)  // K tile via TDM, overlaps the manual V transpose below
      tdm_load_2d((unsigned)(unsigned long long)(void*)&Ks[0][0],
                  Kg + (size_t)kv0 * DMODEL, 64, 64, DK, (unsigned)(SEQ - kv0),
                  DMODEL, 4, 3);
    { // V tile -> Vt transposed (all 128 threads)
      int rowL = tid >> 1;
      int cb   = (tid & 1) * 32;
      const bf16* vs = Vg + (size_t)(kv0 + rowL) * DMODEL + cb;
      v8bf v0 = *(const v8bf*)(vs);
      v8bf v1 = *(const v8bf*)(vs + 8);
      v8bf v2 = *(const v8bf*)(vs + 16);
      v8bf v3 = *(const v8bf*)(vs + 24);
#pragma unroll
      for (int j = 0; j < 8; ++j) {
        Vt[cb + j][rowL]      = v0[j];
        Vt[cb + 8 + j][rowL]  = v1[j];
        Vt[cb + 16 + j][rowL] = v2[j];
        Vt[cb + 24 + j][rowL] = v3[j];
      }
    }
    __builtin_amdgcn_s_wait_tensorcnt(0);
    __syncthreads();

    // S = Q * K^T   (16x64 strip per wave, 2 K-steps of 32 over DK=64)
    v8f sacc[4] = {};
#pragma unroll
    for (int ks = 0; ks < 2; ++ks) {
      int arow = wave * 16 + r;
      v16bf afrag = pack16(*(const v8bf*)(&Qs[arow][ks * 32 + 8 * h]),
                           *(const v8bf*)(&Qs[arow][ks * 32 + 16 + 8 * h]));
#pragma unroll
      for (int ni = 0; ni < 4; ++ni) {
        int bcol = ni * 16 + r;  // kv column
        v16bf bfrag = pack16(*(const v8bf*)(&Ks[bcol][ks * 32 + 8 * h]),
                             *(const v8bf*)(&Ks[bcol][ks * 32 + 16 + 8 * h]));
        sacc[ni] = __builtin_amdgcn_wmma_f32_16x16x32_bf16(
            false, afrag, false, bfrag, (short)0, sacc[ni], false, false);
      }
    }

    // online softmax; row = wave*16 + 8h + i lives in lanes of one half
    const float scale = 0.125f;  // 1/sqrt(DK)
#pragma unroll
    for (int i = 0; i < 8; ++i) {
#pragma unroll
      for (int ni = 0; ni < 4; ++ni) sacc[ni][i] *= scale;
      float mloc = fmaxf(fmaxf(sacc[0][i], sacc[1][i]), fmaxf(sacc[2][i], sacc[3][i]));
#pragma unroll
      for (int m = 1; m < 16; m <<= 1) mloc = fmaxf(mloc, __shfl_xor(mloc, m, 16));
      float mnew = fmaxf(rowm[i], mloc);
      float corr = __expf(rowm[i] - mnew);
      rowm[i] = mnew;
      float psum = 0.f;
#pragma unroll
      for (int ni = 0; ni < 4; ++ni) {
        float p = __expf(sacc[ni][i] - mnew);
        Ps[wave * 16 + 8 * h + i][ni * 16 + r] = f2bf(p);  // wave-private strip
        psum += p;
      }
#pragma unroll
      for (int m = 1; m < 16; m <<= 1) psum += __shfl_xor(psum, m, 16);
      rowl[i] = rowl[i] * corr + psum;
#pragma unroll
      for (int d = 0; d < 4; ++d) accO[d][i] *= corr;
    }

    // O += P * V  (P: 16x64 strip; V via transposed Vt so B-frags are contiguous)
#pragma unroll
    for (int ks = 0; ks < 2; ++ks) {
      int arow = wave * 16 + r;
      v16bf afrag = pack16(*(const v8bf*)(&Ps[arow][ks * 32 + 8 * h]),
                           *(const v8bf*)(&Ps[arow][ks * 32 + 16 + 8 * h]));
#pragma unroll
      for (int d = 0; d < 4; ++d) {
        int bcol = d * 16 + r;  // dk column
        v16bf bfrag = pack16(*(const v8bf*)(&Vt[bcol][ks * 32 + 8 * h]),
                             *(const v8bf*)(&Vt[bcol][ks * 32 + 16 + 8 * h]));
        accO[d] = __builtin_amdgcn_wmma_f32_16x16x32_bf16(
            false, afrag, false, bfrag, (short)0, accO[d], false, false);
      }
    }
    __syncthreads();
  }

  // normalize + write out (bf16, [B,S,D] layout slice)
#pragma unroll
  for (int d = 0; d < 4; ++d)
#pragma unroll
    for (int i = 0; i < 8; ++i) {
      int qrow = wave * 16 + 8 * h + i;
      float inv = 1.0f / rowl[i];
      Og[(size_t)(q0 + qrow) * DMODEL + d * 16 + r] = f2bf(accO[d][i] * inv);
    }
}

// ----------------------------------------------------------------
extern "C" void kernel_launch(void* const* d_in, const int* in_sizes, int n_in,
                              void* d_out, int out_size, void* d_ws, size_t ws_size,
                              hipStream_t stream) {
  const float* emb = (const float*)d_in[0];
  const float* wq  = (const float*)d_in[1];
  const float* wk  = (const float*)d_in[2];
  const float* wv  = (const float*)d_in[3];
  const float* wo  = (const float*)d_in[4];

  char* ws = (char*)d_ws;
  bf16* Xb  = (bf16*)(ws);                        // 16 MiB
  bf16* Wqb = (bf16*)(ws + (16ull << 20));        //  2 MiB
  bf16* Wkb = (bf16*)(ws + (18ull << 20));
  bf16* Wvb = (bf16*)(ws + (20ull << 20));
  bf16* Wob = (bf16*)(ws + (22ull << 20));
  bf16* Qb  = (bf16*)(ws + (24ull << 20));        // 16 MiB each
  bf16* Kb  = (bf16*)(ws + (40ull << 20));
  bf16* Vb  = (bf16*)(ws + (56ull << 20));
  bf16* AOb = (bf16*)(ws + (72ull << 20));        // ends at 88 MiB

  const int M  = BATCH * SEQ;  // 8192
  const int N  = DMODEL;
  const int Kd = DMODEL;

  cvt_f32_bf16<<<512, 256, 0, stream>>>(emb, Xb, (BATCH * SEQ * DMODEL) / 4);
  cvt_f32_bf16<<<256, 256, 0, stream>>>(wq, Wqb, (DMODEL * DMODEL) / 4);
  cvt_f32_bf16<<<256, 256, 0, stream>>>(wk, Wkb, (DMODEL * DMODEL) / 4);
  cvt_f32_bf16<<<256, 256, 0, stream>>>(wv, Wvb, (DMODEL * DMODEL) / 4);
  cvt_f32_bf16<<<256, 256, 0, stream>>>(wo, Wob, (DMODEL * DMODEL) / 4);

  dim3 gg(N / 128, M / 128);
  gemm_xwT<false><<<gg, 256, 0, stream>>>(Xb, Wqb, nullptr, Qb, M, N, Kd);
  gemm_xwT<false><<<gg, 256, 0, stream>>>(Xb, Wkb, nullptr, Kb, M, N, Kd);
  gemm_xwT<false><<<gg, 256, 0, stream>>>(Xb, Wvb, nullptr, Vb, M, N, Kd);

  dim3 ga(SEQ / 64, BATCH * NH);
  flash_attn<<<ga, 128, 0, stream>>>(Qb, Kb, Vb, AOb);

  gemm_xwT<true><<<gg, 256, 0, stream>>>(AOb, Wob, (float*)d_out, nullptr, M, N, Kd);
}
